// Quanv_38714835206268
// MI455X (gfx1250) — matrix operand order repaired and checked by reading, not compiled, not run on hardware
//
#include <hip/hip_runtime.h>

typedef __attribute__((ext_vector_type(16))) _Float16 v16h;
typedef __attribute__((ext_vector_type(8)))  _Float16 v8h;
typedef __attribute__((ext_vector_type(8)))  float    v8f;
typedef __attribute__((ext_vector_type(4)))  float    v4f;

#define BATCH 16
#define CIN   32
#define HIN   224
#define WIN   224
#define OCH   64
#define HOUT  222
#define WOUT  222

#define WTILE 128          // output w positions per block
#define WST   132          // staged input width: 130 needed, padded to 132 for float4 chunks
#define WCHUNKS (WST / 4)  // 33 float4 chunks per (row, channel)
#define CPAD  40           // padded channel stride (f16): 80B -> 16B aligned, conflict-free b128 reads
#define HROWS 2            // output rows per block
#define XROWS (HROWS + 2)  // staged input rows (4)

#define WFRAG_HALFS (4 * 9 * 32 * 16)   // 18432 f16 = 36,864 B of swizzled A-fragments

// ---------------------------------------------------------------------------
// One-time prep: swizzle weights (O,C,9) f32 -> f16 A-fragments in exact
// 16-bit 16x32 A lane/VGPR order (ISA 7.12.2), K ordered as (tap, channel).
// slot = (mt*9 + t)*32 + lane ; element j: K(c) = j + (j>=8?8:0) + (lane>=16?8:0)
// ---------------------------------------------------------------------------
__global__ __launch_bounds__(256)
void prep_weights_kernel(const float* __restrict__ wgt, _Float16* __restrict__ wswz) {
  int f = blockIdx.x * 256 + threadIdx.x;
  if (f >= WFRAG_HALFS) return;
  int j    = f & 15;
  int slot = f >> 4;
  int ln   = slot & 31;
  int rest = slot >> 5;          // mt*9 + t
  int t    = rest % 9;
  int mt   = rest / 9;
  int o    = mt * 16 + (ln & 15);
  int c    = j + ((j >= 8) ? 8 : 0) + ((ln >> 4) * 8);
  int r    = t / 3;
  int s    = t - r * 3;
  wswz[f] = (_Float16)wgt[(o * CIN + c) * 9 + (r * 3 + s)];   // weights: (O, C, 9)
}

// ---------------------------------------------------------------------------
// Implicit-GEMM conv: M = 64 out-chans (4 tiles), N = w positions,
// K = 288 = 9 taps x 32 channels. Each block: 2 output rows x 128 w x all O.
// ---------------------------------------------------------------------------
__global__ __launch_bounds__(256)
void quanv_wmma_kernel(const float* __restrict__ x,
                       const _Float16* __restrict__ wswz,
                       float* __restrict__ out) {
  __shared__ _Float16 s_x[XROWS * WST * CPAD];   // input tile [r][w][c] (42,240 B)
  __shared__ _Float16 s_w[WFRAG_HALFS];          // A fragments          (36,864 B)

  const int tid  = threadIdx.x;
  const int wave = tid >> 5;
  const int lane = tid & 31;

  const int wblk = blockIdx.x;
  const int h0   = blockIdx.y * HROWS;   // 0..220; staged rows h0..h0+3 <= 223, in-bounds
  const int b    = blockIdx.z;
  const int w0   = wblk * WTILE;

  // ---- Async copy of pre-swizzled weights: global -> LDS, no VGPR round-trip ----
  {
    unsigned ldsw = (unsigned)(uintptr_t)(&s_w[0]);
    #pragma unroll
    for (int i = 0; i < 9; ++i) {
      unsigned byteoff = (unsigned)(tid + i * 256) * 16u;   // 2304 x 16B chunks
      asm volatile("global_load_async_to_lds_b128 %0, %1, %2 offset:0"
                   :: "v"(ldsw + byteoff), "v"(byteoff), "s"(wswz)
                   : "memory");
    }
  }

  // ---- Stage input tile as float4 chunks (f32 -> f16), zero-pad beyond width ----
  // Rows are 16B aligned (224*4B pitch, w0 in {0,128}); 224-128=96 % 4 == 0, so
  // edge chunks are all-in or all-out: single per-chunk predicate, no partials.
  for (int f = tid; f < XROWS * CIN * WCHUNKS; f += 256) {
    int w4 = f % WCHUNKS;
    int rc = f / WCHUNKS;
    int c  = rc & (CIN - 1);
    int r  = rc >> 5;
    int gw = w0 + w4 * 4;
    v4f vv = {0.0f, 0.0f, 0.0f, 0.0f};
    if (gw + 3 < WIN)
      vv = *(const v4f*)&x[((b * CIN + c) * HIN + (h0 + r)) * WIN + gw];
    int base = (r * WST + w4 * 4) * CPAD + c;
    s_x[base           ] = (_Float16)vv[0];
    s_x[base +     CPAD] = (_Float16)vv[1];
    s_x[base + 2 * CPAD] = (_Float16)vv[2];
    s_x[base + 3 * CPAD] = (_Float16)vv[3];
  }

  asm volatile("s_wait_asynccnt 0x0" ::: "memory");   // weight copy landed in LDS
  __syncthreads();

  // Each wave: one 16-wide N-tile, both output rows, all four M-tiles.
  const int wbase = wave * 16;
  const int nloc  = lane & 15;
  const int khalf = lane >> 4;   // B layout: lanes 0-15 K=0..7,16..23, lanes 16-31 K=8..15,24..31

  v8f acc[HROWS][4] = {};

  #pragma unroll
  for (int t = 0; t < 9; ++t) {
    const int r = t / 3;
    const int s = t - r * 3;

    // B fragments (32K x 16N) for both output rows: two 8-channel f16 runs (K=8h..8h+7, 16+8h..16+8h+7)
    v16h bfrag[HROWS];
    #pragma unroll
    for (int hs = 0; hs < HROWS; ++hs) {
      const _Float16* bp =
          &s_x[(((hs + r) * WST) + wbase + nloc + s) * CPAD + khalf * 8];
      v8h blo = *(const v8h*)(bp);
      v8h bhi = *(const v8h*)(bp + 16);
      #pragma unroll
      for (int i = 0; i < 8; ++i) { bfrag[hs][i] = blo[i]; bfrag[hs][8 + i] = bhi[i]; }
    }

    #pragma unroll
    for (int mt = 0; mt < 4; ++mt) {
      const v16h afrag = *(const v16h*)&s_w[(((mt * 9) + t) * 32 + lane) * 16];
      #pragma unroll
      for (int hs = 0; hs < HROWS; ++hs) {
        // D = A x B + C : (neg_a, A, neg_b, B, c_mod, C, reuse_a, reuse_b)
        acc[hs][mt] = __builtin_amdgcn_wmma_f32_16x16x32_f16(
            false, afrag, false, bfrag[hs], (short)0, acc[hs][mt], false, false);
      }
    }
  }

  // ---- Epilogue: C/D layout (VGPR v: M=v / v+8 ; N=lane&15) ----
  // out[((o*B + b)*Hn + h)*Wn + w]; non-temporal: output is write-once stream,
  // keep x resident in the 192MB L2 instead.
  const int gw = w0 + wbase + nloc;
  if (gw < WOUT) {
    const int mhalf = khalf * 8;
    #pragma unroll
    for (int hs = 0; hs < HROWS; ++hs) {
      const int gh = h0 + hs;
      #pragma unroll
      for (int mt = 0; mt < 4; ++mt) {
        #pragma unroll
        for (int v = 0; v < 8; ++v) {
          const int o = mt * 16 + mhalf + v;
          __builtin_nontemporal_store(
              acc[hs][mt][v],
              &out[((size_t)(o * BATCH + b) * HOUT + gh) * WOUT + gw]);
        }
      }
    }
  }
}

extern "C" void kernel_launch(void* const* d_in, const int* in_sizes, int n_in,
                              void* d_out, int out_size, void* d_ws, size_t ws_size,
                              hipStream_t stream) {
  (void)in_sizes; (void)n_in; (void)out_size; (void)ws_size;
  const float* x   = (const float*)d_in[0];   // (16, 32, 224, 224) f32
  const float* w   = (const float*)d_in[1];   // (64, 32, 9) f32
  float*       out = (float*)d_out;           // (16, 64, 222, 222) f32, o-major-over-b quirk
  _Float16*    wswz = (_Float16*)d_ws;        // 36,864 B of swizzled f16 A-fragments

  // One-time weight swizzle (72 blocks), then the conv; same stream => ordered.
  prep_weights_kernel<<<dim3((WFRAG_HALFS + 255) / 256), dim3(256), 0, stream>>>(w, wswz);

  dim3 grid((WOUT + WTILE - 1) / WTILE, HOUT / HROWS, BATCH);  // (2, 111, 16)
  quanv_wmma_kernel<<<grid, dim3(256), 0, stream>>>(x, wswz, out);
}
